// kernel_generated_6_62904091017388
// MI455X (gfx1250) — matrix operand order repaired and checked by reading, not compiled, not run on hardware
//
#include <hip/hip_runtime.h>
#include <hip/hip_bf16.h>

// ---------------------------------------------------------------------------
// Shift-conv as tall-skinny GEMM on CDNA5 WMMA (bf16 in, f32 accumulate).
//   out[b,i,n,p] = sum_{l,j,k} V[b,l,j,n,p+k-1] * W[l,j,k,i]
// Stage 0: convert W f32 -> bf16 once into d_ws, pre-swizzled so each wave's
//          B fragment is two contiguous 16B global loads (L2-resident).
// Stage 1: per WG (one batch b, 128 out channels): build V tile in LDS with
//          zero-filled w-halo (9 slots) -> branchless A fragments; K loop has
//          NO barriers and NO staging: 2 vmem + 8 ds loads + 4 wmma per step.
// ---------------------------------------------------------------------------

typedef __bf16 bf16_t;
typedef bf16_t v16bf __attribute__((ext_vector_type(16)));
typedef float  v8f   __attribute__((ext_vector_type(8)));
typedef unsigned int u32x4 __attribute__((ext_vector_type(4)));

#define VT_STRIDE 136   // 128 l-channels + 8 pad halfwords (272B row: 16B aligned, 4-bank skew)
#define VT_ROWS   189   // 3 j * 7 n * 9 q-slots (q = -1..7, halos zero-filled)

union FragAB { u32x4 q[2]; v16bf v; };

__device__ __forceinline__ unsigned short f2bf(float f) {
    union { float f; unsigned u; } c; c.f = f;
    unsigned u = c.u;
    unsigned r = u + 0x7FFFu + ((u >> 16) & 1u);   // round-to-nearest-even
    return (unsigned short)(r >> 16);
}

// ---- Stage 0: W (256,3,3,512) f32 -> bf16, layout [g=jk*8+lb][n(512)][t(32)],
//      where l = lb*32 + t. Each B fragment is then two contiguous 16B chunks.
__global__ __launch_bounds__(256)
void wconvert_bf16(const float* __restrict__ w, unsigned short* __restrict__ wbf,
                   int total) {
    int e = blockIdx.x * 256 + threadIdx.x;
    if (e >= total) return;
    const int t  = e & 31;
    const int n  = (e >> 5) & 511;
    const int g  = e >> 14;          // jk*8 + lb, 0..71
    const int jk = g >> 3;
    const int lb = g & 7;
    const int l  = lb * 32 + t;
    wbf[e] = f2bf(w[(size_t)l * 4608 + jk * 512 + n]);
}

// ---- Stage 1: main GEMM kernel (B fragments straight from pre-swizzled W).
__global__ __launch_bounds__(256)
void shiftconv_wmma_v2(const float* __restrict__ x,
                       const unsigned short* __restrict__ wbf,
                       float* __restrict__ out) {
    __shared__ __align__(16) unsigned short sVt[VT_ROWS * VT_STRIDE]; // ~50.2 KB

    const int tid   = threadIdx.x;
    const int b     = blockIdx.x >> 2;
    const int i0    = (blockIdx.x & 3) << 7;     // N tile base (128 out channels)
    const int wave  = tid >> 5;
    const int lane  = tid & 31;
    const int laneM = lane & 15;
    const int half  = lane >> 4;

    const float* __restrict__ xb = x + (size_t)b * (512 * 49);

    // Per-lane M rows: row index within Vt uses n*9 + p (9 q-slots per n).
    int mrow[4];
#pragma unroll
    for (int m = 0; m < 4; ++m) {
        const int r = m * 16 + laneM;
        mrow[m] = (r / 7) * 9 + (r % 7);
    }
    const bool valid3 = (laneM == 0);            // m=3 tile: only r=48 is real

    // Per-lane base pointer into pre-swizzled bf16 W.
    const unsigned short* __restrict__ wlane =
        wbf + (size_t)(i0 + wave * 16 + laneM) * 32 + half * 8;

    v8f acc[4];
#pragma unroll
    for (int m = 0; m < 4; ++m) acc[m] = (v8f)0.f;

    for (int phase = 0; phase < 2; ++phase) {
        const int lg = phase << 7;               // absolute l base: 0 or 128
        __syncthreads();                         // prior-phase A reads done

        // ---- Build Vt in LDS: rows [j*63 + n*9 + qs], qs=q+1 in [0,8];
        //      qs==0 / qs==8 are zero halos so A loads never need a q-guard.
        for (int e = tid; e < VT_ROWS * 128; e += 256) {
            const int idx = e >> 7;
            const int lo  = e & 127;
            const int j   = idx / 63;
            const int rem = idx - j * 63;
            const int n   = rem / 9;
            const int qs  = rem - n * 9;
            unsigned short val = 0;
            if (qs >= 1 && qs <= 7) {
                const int q = qs - 1;
                int wq = q + 6; if (wq >= 7) wq -= 7;    // W roll (+1)
                const int h1 = n + j - 1;                // group-0 tap
                int n2 = n + 6; if (n2 >= 7) n2 -= 7;
                const int h2 = n2 + j - 1;               // group-1 tap (H-shift)
                const int l = lg + lo;
                float s = 0.f;
                if ((unsigned)h1 < 7u) s += xb[(l * 7 + h1) * 7 + wq];
                if ((unsigned)h2 < 7u) s += xb[((256 + l) * 7 + h2) * 7 + wq];
                val = f2bf(s);
            }
            sVt[idx * VT_STRIDE + lo] = val;
        }
        __syncthreads();

        // ---- K loop: 9 (j,k) taps x 4 l-blocks of 32. No barriers, no staging.
#pragma unroll
        for (int jk = 0; jk < 9; ++jk) {
            const int j  = jk / 3;
            const int kt = jk - j * 3;
            const int rowj = j * 63 + kt;        // + n*9 + p gives qs = p+kt (q=p+kt-1)
#pragma unroll
            for (int lb = 0; lb < 4; ++lb) {
                const int g  = jk * 8 + (phase * 4 + lb);
                const int l0 = lb * 32;

                // B fragment: two contiguous 16B loads from L2-resident bf16 W.
                FragAB bfr;
                const u32x4* pw = (const u32x4*)(wlane + (size_t)g * 16384);
                bfr.q[0] = pw[0];
                bfr.q[1] = pw[2];

                const int aoff = l0 + half * 8;  // K = c..c+7 and c+16..c+23
#pragma unroll
                for (int m = 0; m < 4; ++m) {
                    int row = rowj + mrow[m];
                    if (m == 3) row = valid3 ? row : 0;   // row 0 is a zero halo
                    const unsigned short* pa = &sVt[row * VT_STRIDE + aoff];
                    FragAB af;
                    af.q[0] = *(const u32x4*)pa;
                    af.q[1] = *(const u32x4*)(pa + 16);
                    acc[m] = __builtin_amdgcn_wmma_f32_16x16x32_bf16(
                        false, af.v, false, bfr.v, (short)0, acc[m], false, false);
                }
            }
        }
    }

    // ---- Epilogue: C/D layout -> out[b, ch, n, p]; flat = b*512*49 + ch*49 + r
    const size_t outb = (size_t)b * (512 * 49);
    const int ch = i0 + wave * 16 + laneM;
#pragma unroll
    for (int m = 0; m < 4; ++m) {
#pragma unroll
        for (int v = 0; v < 8; ++v) {
            const int r = m * 16 + v + half * 8;
            if (r < 49) out[outb + (size_t)ch * 49 + r] = acc[m][v];
        }
    }
}

// ---------------------------------------------------------------------------
// Fallback (no workspace needed): stages B tiles through LDS every K step.
// Used only if ws_size is too small for the converted-W buffer.
// ---------------------------------------------------------------------------
#define BT_STRIDE 40

__global__ __launch_bounds__(256)
void shiftconv_wmma_fb(const float* __restrict__ x,
                       const float* __restrict__ w,
                       float* __restrict__ out) {
    __shared__ __align__(16) unsigned short sVt[147 * 136];
    __shared__ __align__(16) unsigned short sBt[128 * BT_STRIDE];

    const int tid   = threadIdx.x;
    const int b     = blockIdx.x >> 2;
    const int i0    = (blockIdx.x & 3) << 7;
    const int wave  = tid >> 5;
    const int lane  = tid & 31;
    const int laneM = lane & 15;
    const int half  = lane >> 4;

    const float* __restrict__ xb = x + (size_t)b * (512 * 49);

    int rr[4], pp[4];
#pragma unroll
    for (int m = 0; m < 4; ++m) { rr[m] = m * 16 + laneM; pp[m] = rr[m] % 7; }

    v8f acc[4];
#pragma unroll
    for (int m = 0; m < 4; ++m) acc[m] = (v8f)0.f;

    for (int phase = 0; phase < 2; ++phase) {
        const int lg = phase << 7;
        __syncthreads();
        for (int e = tid; e < 147 * 128; e += 256) {
            const int idx = e >> 7, lo = e & 127;
            const int j = idx / 49, rem = idx - j * 49;
            const int n = rem / 7,  q = rem - n * 7;
            int wq = q + 6; if (wq >= 7) wq -= 7;
            const int h1 = n + j - 1;
            int n2 = n + 6; if (n2 >= 7) n2 -= 7;
            const int h2 = n2 + j - 1;
            const int l = lg + lo;
            float s = 0.f;
            if ((unsigned)h1 < 7u) s += xb[(l * 7 + h1) * 7 + wq];
            if ((unsigned)h2 < 7u) s += xb[((256 + l) * 7 + h2) * 7 + wq];
            sVt[idx * 136 + lo] = f2bf(s);
        }
        for (int jk = 0; jk < 9; ++jk) {
            const int j = jk / 3, kt = jk - j * 3;
            const float* __restrict__ wbase = w + (size_t)jk * 512 + i0;
            for (int l0 = 0; l0 < 128; l0 += 32) {
                __syncthreads();
                for (int e = tid; e < 32 * 128; e += 256) {
                    const int t = e >> 7, nn = e & 127;
                    sBt[nn * BT_STRIDE + t] =
                        f2bf(wbase[(size_t)(lg + l0 + t) * 4608 + nn]);
                }
                __syncthreads();
                FragAB bfr;
                {
                    const unsigned short* pB =
                        &sBt[(wave * 16 + laneM) * BT_STRIDE + half * 8];
                    bfr.q[0] = *(const u32x4*)pB;
                    bfr.q[1] = *(const u32x4*)(pB + 16);
                }
#pragma unroll
                for (int m = 0; m < 4; ++m) {
                    const int q = pp[m] + kt - 1;
                    const bool valid = (rr[m] < 49) && ((unsigned)q < 7u);
                    FragAB af;
                    if (valid) {
                        const unsigned short* pa =
                            &sVt[(j * 49 + rr[m] + kt - 1) * 136 + l0 + half * 8];
                        af.q[0] = *(const u32x4*)pa;
                        af.q[1] = *(const u32x4*)(pa + 16);
                    } else {
                        af.q[0] = (u32x4)0u;
                        af.q[1] = (u32x4)0u;
                    }
                    acc[m] = __builtin_amdgcn_wmma_f32_16x16x32_bf16(
                        false, af.v, false, bfr.v, (short)0, acc[m], false, false);
                }
            }
        }
    }

    const size_t outb = (size_t)b * (512 * 49);
    const int ch = i0 + wave * 16 + laneM;
#pragma unroll
    for (int m = 0; m < 4; ++m) {
#pragma unroll
        for (int v = 0; v < 8; ++v) {
            const int r = m * 16 + v + half * 8;
            if (r < 49) out[outb + (size_t)ch * 49 + r] = acc[m][v];
        }
    }
}

extern "C" void kernel_launch(void* const* d_in, const int* in_sizes, int n_in,
                              void* d_out, int out_size, void* d_ws, size_t ws_size,
                              hipStream_t stream) {
    const float* x = (const float*)d_in[0];   // (1024, 512, 7, 7) f32
    const float* w = (const float*)d_in[1];   // (256, 3, 3, 512) f32
    float* out = (float*)d_out;               // (1024, 512, 7, 7) f32

    const int wtot = in_sizes[1];             // 1,179,648
    const int Bn   = in_sizes[0] / (512 * 49);
    const size_t need = (size_t)wtot * sizeof(unsigned short);

    if (ws_size >= need) {
        unsigned short* wbf = (unsigned short*)d_ws;
        wconvert_bf16<<<(wtot + 255) / 256, 256, 0, stream>>>(w, wbf, wtot);
        shiftconv_wmma_v2<<<Bn * 4, 256, 0, stream>>>(x, wbf, out);
    } else {
        shiftconv_wmma_fb<<<Bn * 4, 256, 0, stream>>>(x, w, out);
    }
}
